// OptimalEmpiricalDenoiserConstantEnergy_88287347736857
// MI455X (gfx1250) — matrix operand order repaired and checked by reading, not compiled, not run on hardware
//
#include <hip/hip_runtime.h>
#include <hip/hip_bf16.h>

typedef __bf16 bf16_t;
typedef __attribute__((ext_vector_type(16))) __bf16 v16bf;
typedef __attribute__((ext_vector_type(8)))  float  v8f;

#define NQ     16      // query rows per block (N tile)
#define DF     3072    // feature dim (C*H*W)
#define MC     32      // dataset rows per chunk (= WMMA K of GEMM2)
#define NWAVE  16
#define TPB    512     // 16 wave32
#define KSLICE 384     // GEMM1 K-slice per wave (DF/8 k-groups)
#define WCOL   192     // output columns owned per wave (DF/NWAVE)
#define NOT    12      // WCOL/16 accumulator tiles per wave
#define SCL    4.0f    // 1 / NOISE_VAR
#define NEGBIG (-3.0e38f)

struct SMem {
  uint4  xbq[NQ][DF / 8];        // 96KB : X tile, bf16 pairs, row-major (WMMA A source)
  uint4  dtrq[DF][4];            // 192KB: chunk, bf16, transposed [col][4 x 16B row-blocks], XOR-swizzled
  float  s4[8][NQ][MC];          // 16KB : per-k-group partial scores (deterministic reduce)
  uint4  pq[NQ][4];              // 1KB  : softmax weights bf16 (WMMA A source for GEMM2)
  float  mrow[NQ], lrow[NQ], arow[NQ];
};

__device__ __forceinline__ unsigned int pkbf(float a, float b) {
  union { bf16_t h[2]; unsigned int u; } z;
  z.h[0] = (bf16_t)a; z.h[1] = (bf16_t)b;
  return z.u;
}

union ABFrag { uint4 q[2]; v16bf v; };

__global__ __launch_bounds__(TPB, 1)
void denoise_main(const float* __restrict__ X, const float* __restrict__ Dg,
                  float* __restrict__ opart, float* __restrict__ stats,
                  int Mtot, int Mper)
{
  __shared__ SMem sm;
  const int tid   = threadIdx.x;
  const int w     = tid >> 5;      // wave 0..15
  const int lane  = tid & 31;
  const int hi    = lane >> 4;     // lane half (WMMA layout)
  const int ln    = lane & 15;
  const int split = blockIdx.x;
  const int ntile = blockIdx.y;

  // ---------------- preload X tile as bf16 into LDS (one b128 store per 8 floats) ----
  {
    const float4* xg = (const float4*)(X + (size_t)ntile * NQ * DF);
    uint4* xb = (uint4*)sm.xbq;
#pragma unroll
    for (int it = 0; it < (NQ * DF / 8) / TPB; ++it) {
      const int e = it * TPB + tid;       // uint4 index == 8 consecutive floats
      const float4 f0 = xg[e * 2 + 0];
      const float4 f1 = xg[e * 2 + 1];
      uint4 q;
      q.x = pkbf(f0.x, f0.y); q.y = pkbf(f0.z, f0.w);
      q.z = pkbf(f1.x, f1.y); q.w = pkbf(f1.z, f1.w);
      xb[e] = q;
    }
  }
  if (tid < NQ) { sm.mrow[tid] = NEGBIG; sm.lrow[tid] = 0.f; }

  v8f o[NOT];
#pragma unroll
  for (int i = 0; i < NOT; ++i) o[i] = (v8f){0.f,0.f,0.f,0.f,0.f,0.f,0.f,0.f};

  const int mStart = split * Mper;
  const int mEnd0  = (Mtot < mStart + Mper) ? Mtot : (mStart + Mper);

  __syncthreads();

  for (int m0 = mStart; m0 < mEnd0; m0 += MC) {
    const int nv = (mEnd0 - m0) < MC ? (mEnd0 - m0) : MC;

    // ------- stage chunk: global fp32 -> bf16, transposed + swizzled LDS -------
    for (int blk = w; blk < DF / 32; blk += NWAVE) {
      const int c  = blk * 32 + lane;        // feature column owned by this lane
      const int sw = (c >> 2) & 3;           // XOR swizzle for bank-conflict-free b128
#pragma unroll
      for (int jb = 0; jb < 4; ++jb) {       // 8 chunk-rows per 16B block
        float vv[8];
#pragma unroll
        for (int j2 = 0; j2 < 8; ++j2) {
          const int j = jb * 8 + j2;
          vv[j2] = (j < nv) ? Dg[(size_t)(m0 + j) * DF + c] : 0.f;
        }
        uint4 q;
        q.x = pkbf(vv[0], vv[1]); q.y = pkbf(vv[2], vv[3]);
        q.z = pkbf(vv[4], vv[5]); q.w = pkbf(vv[6], vv[7]);
        sm.dtrq[c][jb ^ sw] = q;
        if (m0 + MC < mEnd0) {               // warm L2 for next chunk
          int pr = m0 + MC + jb * 8; if (pr >= Mtot) pr = Mtot - 1;
          __builtin_prefetch(&Dg[(size_t)pr * DF + c], 0, 1);
        }
      }
    }
    __syncthreads();

    // ------- GEMM1: S[16xMC] = X · Dchunk^T  (wave: tile t, k-group g) -------
    {
      const int  t      = w & 1;
      const int  g      = w >> 1;
      const int  mg     = m0 + t * 16 + ln;  // dataset row feeding this lane's B column
      const bool mvalid = (mg < mEnd0);
      const int  kbase  = g * KSLICE;
      v8f sacc = (v8f){0.f,0.f,0.f,0.f,0.f,0.f,0.f,0.f};
#pragma unroll
      for (int kk = 0; kk < KSLICE; kk += 32) {
        const int k0 = kbase + kk;
        ABFrag a;    // A fragment from LDS X tile: b128 x2 with folded offsets
        a.q[0] = sm.xbq[ln][k0 / 8 + 2 * hi];
        a.q[1] = sm.xbq[ln][k0 / 8 + 2 * hi + 1];
        const float4* bp = (const float4*)(Dg + (size_t)mg * DF + k0 + hi * 16);
        const float4 z = {0.f, 0.f, 0.f, 0.f};
        const float4 f0 = mvalid ? bp[0] : z;
        const float4 f1 = mvalid ? bp[1] : z;
        const float4 f2 = mvalid ? bp[2] : z;
        const float4 f3 = mvalid ? bp[3] : z;
        union { unsigned int u[8]; v16bf v; } b;
        b.u[0] = pkbf(f0.x, f0.y); b.u[1] = pkbf(f0.z, f0.w);
        b.u[2] = pkbf(f1.x, f1.y); b.u[3] = pkbf(f1.z, f1.w);
        b.u[4] = pkbf(f2.x, f2.y); b.u[5] = pkbf(f2.z, f2.w);
        b.u[6] = pkbf(f3.x, f3.y); b.u[7] = pkbf(f3.z, f3.w);
        sacc = __builtin_amdgcn_wmma_f32_16x16x32_bf16(false, a.v, false, b.v,
                                                       (short)0, sacc, false, false);
      }
#pragma unroll
      for (int r = 0; r < 8; ++r)
        sm.s4[g][r + hi * 8][t * 16 + ln] = sacc[r];
    }
    __syncthreads();

    // ------- online softmax: wave w owns query row w, lane = chunk column -------
    {
      float s = 0.f;
#pragma unroll
      for (int g = 0; g < 8; ++g) s += sm.s4[g][w][lane];   // fixed-order -> deterministic
      if (m0 + lane >= mEnd0) s = NEGBIG;
      float mx = s;
#pragma unroll
      for (int off = 16; off > 0; off >>= 1)
        mx = fmaxf(mx, __shfl_xor(mx, off, 32));
      const float mold = sm.mrow[w];
      const float mnew = fmaxf(mold, mx);
      const float pv   = __expf((s - mnew) * SCL);
      const bf16_t ph  = (bf16_t)pv;
      float ls = (float)ph;                  // l consistent with bf16-rounded weights
#pragma unroll
      for (int off = 16; off > 0; off >>= 1)
        ls += __shfl_xor(ls, off, 32);
      ((bf16_t*)sm.pq)[w * MC + lane] = ph;
      if (lane == 0) {
        const float alpha = __expf((mold - mnew) * SCL);
        sm.arow[w] = alpha;
        sm.lrow[w] = sm.lrow[w] * alpha + ls;
        sm.mrow[w] = mnew;
      }
    }
    __syncthreads();

    // ------- rescale O and GEMM2: O += P · Dchunk -------
    {
      float ar[8];
#pragma unroll
      for (int r = 0; r < 8; ++r) ar[r] = sm.arow[r + hi * 8];
#pragma unroll
      for (int tIdx = 0; tIdx < NOT; ++tIdx)
#pragma unroll
        for (int r = 0; r < 8; ++r) o[tIdx][r] *= ar[r];

      ABFrag pa;   // A fragment = softmax weights (b128 x2)
      pa.q[0] = sm.pq[ln][2 * hi];
      pa.q[1] = sm.pq[ln][2 * hi + 1];
      const int jb0 = hi * 2;
      const int swb = ((w * WCOL + ln) >> 2) & 3;   // swizzle invariant across tiles (step 16)
#pragma unroll
      for (int tIdx = 0; tIdx < NOT; ++tIdx) {
        const int c = w * WCOL + tIdx * 16 + ln;
        ABFrag b;  // B fragment from transposed/swizzled chunk: b128 x2
        b.q[0] = sm.dtrq[c][(jb0 + 0) ^ swb];
        b.q[1] = sm.dtrq[c][(jb0 + 1) ^ swb];
        o[tIdx] = __builtin_amdgcn_wmma_f32_16x16x32_bf16(false, pa.v, false, b.v,
                                                          (short)0, o[tIdx], false, false);
      }
    }
    __syncthreads();
  }

  // ------- write partial O (unnormalized) and per-row (m, l) stats -------
  float* od = opart + (size_t)split * 256 * DF + (size_t)ntile * NQ * DF;
#pragma unroll
  for (int tIdx = 0; tIdx < NOT; ++tIdx) {
    const int c = w * WCOL + tIdx * 16 + ln;
#pragma unroll
    for (int r = 0; r < 8; ++r)
      od[(size_t)(r + hi * 8) * DF + c] = o[tIdx][r];
  }
  if (tid < NQ) {
    const int grow = ntile * NQ + tid;
    stats[((size_t)split * 256 + grow) * 2 + 0] = sm.mrow[tid] * SCL;
    stats[((size_t)split * 256 + grow) * 2 + 1] = sm.lrow[tid];
  }
}

// ---- combine split partials (fixed-order => deterministic), or normalize in-place ----
__global__ void denoise_combine(const float* __restrict__ part,
                                const float* __restrict__ stats,
                                float* __restrict__ outp,
                                int nsplit, int direct)
{
  const int row = blockIdx.x;     // 0..255
  const int tid = threadIdx.x;    // 256
  if (direct) {
    const float inv = 1.f / stats[row * 2 + 1];
    for (int c = tid; c < DF; c += 256)
      outp[(size_t)row * DF + c] *= inv;
    return;
  }
  float mstar = NEGBIG;
  for (int s = 0; s < nsplit; ++s)
    mstar = fmaxf(mstar, stats[((size_t)s * 256 + row) * 2]);
  float L = 0.f;
  for (int s = 0; s < nsplit; ++s)
    L += __expf(stats[((size_t)s * 256 + row) * 2] - mstar) *
         stats[((size_t)s * 256 + row) * 2 + 1];
  const float invL = 1.f / L;
  for (int c = tid; c < DF; c += 256) {
    float acc = 0.f;
    for (int s = 0; s < nsplit; ++s) {
      const float sc = __expf(stats[((size_t)s * 256 + row) * 2] - mstar);
      acc += sc * part[((size_t)s * 256 + row) * DF + c];
    }
    outp[(size_t)row * DF + c] = acc * invL;
  }
}

extern "C" void kernel_launch(void* const* d_in, const int* in_sizes, int n_in,
                              void* d_out, int out_size, void* d_ws, size_t ws_size,
                              hipStream_t stream)
{
  const float* x  = (const float*)d_in[0];
  const float* ds = (const float*)d_in[1];
  float* outp = (float*)d_out;
  const int Mtot   = in_sizes[1] / DF;   // 50000
  const int Nrow   = in_sizes[0] / DF;   // 256
  const int ntiles = Nrow / NQ;          // 16

  const size_t statsCap = (size_t)32 * 256 * 2 * sizeof(float);
  float* stats = (float*)d_ws;
  float* part  = (float*)((char*)d_ws + statsCap);

  int split = 0;
  const int cand[5] = {32, 16, 8, 4, 2};
  for (int i = 0; i < 5 && !split; ++i) {
    const size_t need = statsCap + (size_t)cand[i] * Nrow * DF * sizeof(float);
    if (ws_size >= need) split = cand[i];
  }
  int direct = 0;
  if (!split) {
    if (ws_size >= statsCap + (size_t)Nrow * DF * sizeof(float)) split = 1;
    else { split = 1; direct = 1; }          // write O straight to d_out, normalize in place
  }
  const int Mper = ((Mtot + split - 1) / split + MC - 1) & ~(MC - 1);

  float* opart = direct ? outp : part;
  denoise_main<<<dim3(split, ntiles), TPB, 0, stream>>>(x, ds, opart, stats, Mtot, Mper);
  denoise_combine<<<dim3(Nrow), 256, 0, stream>>>(part, stats, outp, split, direct);
}
